// ChebyNetKAN_3_48137993453858
// MI455X (gfx1250) — matrix operand order repaired and checked by reading, not compiled, not run on hardware
//
#include <hip/hip_runtime.h>

// ---------------------------------------------------------------------------
// ChebyNetKAN forward for MI455X (gfx1250, wave32, WMMA bf16 path,
// async global->LDS staging with double buffering)
// ---------------------------------------------------------------------------

typedef __attribute__((ext_vector_type(16))) __bf16 v16bf;
typedef __attribute__((ext_vector_type(8)))  float  v8f;
typedef int v4i128 __attribute__((vector_size(16)));

#define NNODES 20000
#define NBR    8
#define FEATD  256
#define HIDD   512
#define CATD   4096
#define F1D    1024
#define K1D    32
#define OUTD   10

// ---------------------------------------------------------------------------
// gfx1250 async global->LDS copy (GLOBAL_LOAD_ASYNC_TO_LDS_B128, ASYNCcnt)
// ---------------------------------------------------------------------------
#if defined(__AMDGCN__) && __has_builtin(__builtin_amdgcn_global_load_async_to_lds_b128)
#define HAVE_ASYNC_LDS 1
#else
#define HAVE_ASYNC_LDS 0
#endif

__device__ __forceinline__ void async_cp16(void* l, const void* g) {
#if HAVE_ASYNC_LDS
  __builtin_amdgcn_global_load_async_to_lds_b128(
      (__attribute__((address_space(1))) v4i128*)g,
      (__attribute__((address_space(3))) v4i128*)l, 0, 0);
#else
  *(uint4*)l = *(const uint4*)g;
#endif
}

template <int N>
__device__ __forceinline__ void wait_async_lds() {
#if HAVE_ASYNC_LDS
#if __has_builtin(__builtin_amdgcn_s_wait_asynccnt)
  __builtin_amdgcn_s_wait_asynccnt(N);
#else
  asm volatile("s_wait_asynccnt %0" :: "i"(N) : "memory");
#endif
#endif
}

// ---------------------------------------------------------------------------
// Closed-form cardinal cubic B-spline on uniform knots (h = 0.4, range ±2.2).
// basis_j(x) = b3((x+2.2)/0.4 - j),  j = 0..7   (matches Cox-de Boor order 3)
// ---------------------------------------------------------------------------
__device__ __forceinline__ float bspl3(float s) {
  if (s <= 0.f || s >= 4.f) return 0.f;
  if (s < 1.f) return s * s * s * (1.f / 6.f);
  if (s < 2.f) return (s * (s * (-3.f * s + 12.f) - 12.f) + 4.f) * (1.f / 6.f);
  if (s < 3.f) return (s * (s * (3.f * s - 24.f) + 60.f) - 44.f) * (1.f / 6.f);
  float q = 4.f - s;
  return q * q * q * (1.f / 6.f);
}

// ---------------------------------------------------------------------------
// WMMA operand fragment loaders from row-major LDS tiles (pitch in elements,
// (pitch*2) must be a multiple of 16 bytes).
// A operand (16x32 bf16): lane L -> row L&15; K = kb..kb+7 and kb+16..kb+23,
//                         kb = (L>>4)*8.
// B operand (32x16 bf16, stored N-major): lane L -> col L&15; K = kb..kb+7
//                         and kb+16..kb+23, kb = (L>>4)*8.
// ---------------------------------------------------------------------------
__device__ __forceinline__ v16bf frag_a(const __bf16* base, int pitch, int lane, int ks) {
  const __bf16* p = base + (lane & 15) * pitch + ks + ((lane >> 4) << 3);
  union { v16bf v; uint4 u[2]; } f;
  f.u[0] = *(const uint4*)(p);
  f.u[1] = *(const uint4*)(p + 16);
  return f.v;
}
__device__ __forceinline__ v16bf frag_b(const __bf16* base, int pitch, int lane, int ks) {
  const __bf16* p = base + (lane & 15) * pitch + ks + ((lane >> 4) << 3);
  union { v16bf v; uint4 u[2]; } f;
  f.u[0] = *(const uint4*)(p);
  f.u[1] = *(const uint4*)(p + 16);
  return f.v;
}

// ---------------------------------------------------------------------------
// Elementwise f32 -> bf16
// ---------------------------------------------------------------------------
__global__ void cvt_bf16_kernel(const float* __restrict__ src, __bf16* __restrict__ dst, long n) {
  long i = (long)blockIdx.x * blockDim.x + threadIdx.x;
  if (i < n) dst[i] = (__bf16)src[i];
}

// ---------------------------------------------------------------------------
// Batched transpose + convert:  src [batch][K][N] f32 -> dst [batch][N][K] bf16
// ---------------------------------------------------------------------------
__global__ void transpose_cvt_kernel(const float* __restrict__ src, __bf16* __restrict__ dst,
                                     int batch, int K, int Nn) {
  long i = (long)blockIdx.x * blockDim.x + threadIdx.x;
  long total = (long)batch * K * Nn;
  if (i >= total) return;
  int b = (int)(i / ((long)K * Nn));
  long rem = i - (long)b * K * Nn;
  int k = (int)(rem / Nn);
  int n = (int)(rem - (long)k * Nn);
  dst[(long)b * Nn * K + (long)n * K + k] = (__bf16)src[i];
}

// ---------------------------------------------------------------------------
// Build combined KAN-1 weight, bf16, layout [32][9*1024] with k = c*1024 + i:
//   c==0 -> base_w[o][i];  c>0 -> spline_w[o][i][c-1] * scaler[o][i]
// ---------------------------------------------------------------------------
__global__ void build_wk1_kernel(const float* __restrict__ bw1, const float* __restrict__ sw1,
                                 const float* __restrict__ sc1, __bf16* __restrict__ wk1t) {
  int i = blockIdx.x * blockDim.x + threadIdx.x;
  if (i >= K1D * 9 * F1D) return;
  int o  = i / (9 * F1D);
  int k  = i - o * (9 * F1D);
  int c  = k >> 10;         // / 1024
  int ii = k & 1023;
  float v = (c == 0) ? bw1[o * F1D + ii]
                     : sw1[(o * F1D + ii) * 8 + (c - 1)] * sc1[o * F1D + ii];
  wk1t[i] = (__bf16)v;
}

// ---------------------------------------------------------------------------
// Stage one K-tile (A: 64x64, B: 128x64 bf16) into LDS. 6 x b128 per thread
// (= 6 async ops per wave on the async path).
// ---------------------------------------------------------------------------
__device__ __forceinline__ void gemm_stage(
    __bf16 (*dA)[72], __bf16 (*dB)[72],
    const __bf16* __restrict__ A, long aBlockStride, int kBlock, int lda,
    const __bf16* __restrict__ Bt, int ldbt,
    int m0, int n0, int M, int k0, int tid)
{
  const int kb = k0 / kBlock;
  const int kr = k0 - kb * kBlock;
  const __bf16* Ak = A + (long)kb * aBlockStride + kr;
#if HAVE_ASYNC_LDS
#pragma unroll
  for (int j = 0; j < 2; j++) {
    int idx = tid + (j << 8);
    int r = idx >> 3, cg = idx & 7;
    int m = m0 + r; if (m >= M) m = M - 1;   // row-clamp (rows independent)
    async_cp16(&dA[r][cg << 3], Ak + (long)m * lda + (cg << 3));
  }
#pragma unroll
  for (int j = 0; j < 4; j++) {
    int idx = tid + (j << 8);
    int n = idx >> 3, cg = idx & 7;
    async_cp16(&dB[n][cg << 3], Bt + (long)(n0 + n) * ldbt + k0 + (cg << 3));
  }
#else
  // batched synchronous fallback: all loads in flight before first LDS store
  uint4 ra[2], rb[4];
#pragma unroll
  for (int j = 0; j < 2; j++) {
    int idx = tid + (j << 8);
    int cg = idx & 7;
    int m = m0 + (idx >> 3); if (m >= M) m = M - 1;
    ra[j] = *(const uint4*)(Ak + (long)m * lda + (cg << 3));
  }
#pragma unroll
  for (int j = 0; j < 4; j++) {
    int idx = tid + (j << 8);
    int n = idx >> 3, cg = idx & 7;
    rb[j] = *(const uint4*)(Bt + (long)(n0 + n) * ldbt + k0 + (cg << 3));
  }
#pragma unroll
  for (int j = 0; j < 2; j++) {
    int idx = tid + (j << 8);
    *(uint4*)(&dA[idx >> 3][(idx & 7) << 3]) = ra[j];
  }
#pragma unroll
  for (int j = 0; j < 4; j++) {
    int idx = tid + (j << 8);
    *(uint4*)(&dB[idx >> 3][(idx & 7) << 3]) = rb[j];
  }
#endif
}

// ---------------------------------------------------------------------------
// Batched bf16 WMMA GEMM:  C[m][n] = sum_k A[m][k] * Bt[n][k]  (+bias, +relu)
// A addressing supports the concat view: element (m,k) lives at
//   A + kb*aBlockStride + m*lda + (k - kb*kBlock),  kb = k / kBlock.
// Macro tile 64x128, 8 waves (4 M-subtiles x 2 N-halves), K-tile 64,
// double-buffered LDS with async prefetch of the next tile.
// ---------------------------------------------------------------------------
__global__ __launch_bounds__(256) void gemm_bf16_kernel(
    const __bf16* __restrict__ A, long aBatch, int lda, int kBlock, long aBlockStride,
    const __bf16* __restrict__ Bt, long bBatch, int ldbt,
    const float* __restrict__ bias, long biasBatch,
    float* __restrict__ C, long cBatch, int ldc,
    int M, int K, int relu)
{
  __shared__ __bf16 sA[2][64][72];
  __shared__ __bf16 sB[2][128][72];

  const int tid  = threadIdx.x;
  const int lane = tid & 31;
  const int wave = tid >> 5;
  const int mw   = wave & 3;   // 16-row subtile
  const int nw   = wave >> 2;  // 64-col half
  const int m0   = blockIdx.x * 64;
  const int n0   = blockIdx.y * 128;
  const int bz   = blockIdx.z;

  A    += (long)bz * aBatch;
  Bt   += (long)bz * bBatch;
  C    += (long)bz * cBatch;
  bias += (long)bz * biasBatch;

  v8f acc[4];
#pragma unroll
  for (int t = 0; t < 4; t++) acc[t] = (v8f){0.f, 0.f, 0.f, 0.f, 0.f, 0.f, 0.f, 0.f};

  const int nIter = K >> 6;   // K / 64
  gemm_stage(sA[0], sB[0], A, aBlockStride, kBlock, lda, Bt, ldbt, m0, n0, M, 0, tid);

  for (int it = 0; it < nIter; ++it) {
    const int cur = it & 1;
    if (it + 1 < nIter) {
      gemm_stage(sA[cur ^ 1], sB[cur ^ 1], A, aBlockStride, kBlock, lda,
                 Bt, ldbt, m0, n0, M, (it + 1) << 6, tid);
      wait_async_lds<6>();    // previous tile's 6 per-wave async ops retired
    } else {
      wait_async_lds<0>();
    }
    __syncthreads();

#pragma unroll
    for (int ks = 0; ks < 64; ks += 32) {
      v16bf a = frag_a(&sA[cur][mw << 4][0], 72, lane, ks);
      v16bf b[4];
#pragma unroll
      for (int t = 0; t < 4; t++)
        b[t] = frag_b(&sB[cur][(nw << 6) + (t << 4)][0], 72, lane, ks);
#pragma unroll
      for (int t = 0; t < 4; t++)
        acc[t] = __builtin_amdgcn_wmma_f32_16x16x32_bf16(
            false, a, false, b[t], (short)0, acc[t], false, false);
    }
    __syncthreads();          // all waves done reading before buffer refilled
  }

  // --- epilogue: C/D layout -> lane L: col = L&15; VGPR j: row = j + 8*(L>>4)
  const int rowBase = m0 + (mw << 4) + ((lane >> 4) << 3);
  const int colBase = n0 + (nw << 6) + (lane & 15);
#pragma unroll
  for (int t = 0; t < 4; t++) {
    const int col = colBase + (t << 4);
    const float bv = bias ? bias[col] : 0.f;
#pragma unroll
    for (int j = 0; j < 8; j++) {
      int row = rowBase + j;
      if (row < M) {
        float v = acc[t][j] + bv;
        if (relu) v = v > 0.f ? v : 0.f;
        C[(long)row * ldc + col] = v;
      }
    }
  }
}

// ---------------------------------------------------------------------------
// Column statistics (deterministic two-stage): partial sums over row chunks.
// h: [NBR][rows][Cper] f32; gc = b*Cper + ch; psum/psumsq: [chunks][NBR*Cper]
// ---------------------------------------------------------------------------
__global__ void colstats_kernel(const float* __restrict__ h,
                                float* __restrict__ psum, float* __restrict__ psumsq,
                                int rows, int rowsPer, int Ctot, int Cper) {
  int gc = blockIdx.x * blockDim.x + threadIdx.x;
  if (gc >= Ctot) return;
  int b  = gc / Cper;
  int ch = gc - b * Cper;
  long base = (long)b * rows * Cper + ch;
  int r0 = blockIdx.y * rowsPer;
  int r1 = r0 + rowsPer; if (r1 > rows) r1 = rows;
  float s = 0.f, s2 = 0.f;
  for (int r = r0; r < r1; r++) {
    float v = h[base + (long)r * Cper];
    s += v; s2 += v * v;
  }
  psum  [(long)blockIdx.y * Ctot + gc] = s;
  psumsq[(long)blockIdx.y * Ctot + gc] = s2;
}

__global__ void colstats_finalize_kernel(const float* __restrict__ psum,
                                         const float* __restrict__ psumsq,
                                         float* __restrict__ sums, float* __restrict__ sumsq,
                                         int Ctot, int chunks) {
  int gc = blockIdx.x * blockDim.x + threadIdx.x;
  if (gc >= Ctot) return;
  float s = 0.f, s2 = 0.f;
  for (int c = 0; c < chunks; c++) {
    s  += psum  [(long)c * Ctot + gc];
    s2 += psumsq[(long)c * Ctot + gc];
  }
  sums[gc] = s; sumsq[gc] = s2;
}

// ---------------------------------------------------------------------------
// Fused batchnorm (training stats) + ReLU + f32->bf16
// ---------------------------------------------------------------------------
__global__ void bn_relu_bf16_kernel(const float* __restrict__ h,
                                    const float* __restrict__ sums, const float* __restrict__ sumsq,
                                    const float* __restrict__ gamma, const float* __restrict__ beta,
                                    __bf16* __restrict__ out, int rows, int Cper, float invN) {
  long i = (long)blockIdx.x * blockDim.x + threadIdx.x;
  long total = (long)NBR * rows * Cper;
  if (i >= total) return;
  int ch = (int)(i % Cper);
  int b  = (int)(i / ((long)rows * Cper));
  int gc = b * Cper + ch;
  float mu  = sums[gc] * invN;
  float var = sumsq[gc] * invN - mu * mu;
  float v = gamma[gc] * (h[i] - mu) * rsqrtf(var + 1e-5f) + beta[gc];
  out[i] = (__bf16)(v > 0.f ? v : 0.f);
}

// ---------------------------------------------------------------------------
// KAN layer 1, fused expansion + WMMA GEMM.
// f:[NNODES][1024] f32 -> out1:[NNODES][32] f32, using wk1t:[32][9*1024] bf16.
// Block: 128 thr = 4 waves, tile 64 rows x 32 cols; per 32-input chunk the 9
// expansion channels become 9 bf16 A tiles in LDS, then 9x2 WMMAs.
// ---------------------------------------------------------------------------
__global__ __launch_bounds__(128) void kan1_kernel(const float* __restrict__ f,
                                                   const __bf16* __restrict__ wk1t,
                                                   float* __restrict__ out1) {
  __shared__ __bf16 sA9[9][64 * 32];   // [c][r*32 + ii]
  __shared__ __bf16 sB9[9][32 * 32];   // [c][o*32 + ii]

  const int tid  = threadIdx.x;
  const int lane = tid & 31;
  const int wave = tid >> 5;           // 0..3 -> 16-row subtile
  const int m0   = blockIdx.x * 64;

  v8f acc[2];
  acc[0] = (v8f){0.f, 0.f, 0.f, 0.f, 0.f, 0.f, 0.f, 0.f};
  acc[1] = (v8f){0.f, 0.f, 0.f, 0.f, 0.f, 0.f, 0.f, 0.f};

  for (int i0 = 0; i0 < F1D; i0 += 32) {
    // --- async-stage weights: 9 x 32 rows x 32 k = 1152 b128, 9 per thread
#pragma unroll
    for (int j = 0; j < 9; j++) {
      int idx = tid + (j << 7);        // 0..1151
      int c   = idx >> 7;              // 0..8
      int rem = idx & 127;
      int n   = rem >> 2;              // output col 0..31
      int cg  = rem & 3;
      async_cp16(&sB9[c][(n << 5) + (cg << 3)],
                 wk1t + (long)n * (9 * F1D) + c * F1D + i0 + (cg << 3));
    }
    // --- expansion: 64x32 inputs -> 9 channels, 16 elements per thread
#pragma unroll 4
    for (int j = 0; j < 16; j++) {
      int e  = tid + (j << 7);
      int r  = e >> 5;
      int ii = e & 31;
      int row = m0 + r;
      float x = (row < NNODES) ? f[(long)row * F1D + i0 + ii] : 0.f;
      float sig = 1.f / (1.f + __expf(-x));
      sA9[0][(r << 5) + ii] = (__bf16)(x * sig);          // SiLU channel
      float u = (x + 2.2f) * 2.5f;
#pragma unroll
      for (int c = 0; c < 8; c++)
        sA9[c + 1][(r << 5) + ii] = (__bf16)bspl3(u - (float)c);
    }
    wait_async_lds<0>();
    __syncthreads();

#pragma unroll
    for (int c = 0; c < 9; c++) {
      v16bf a = frag_a(&sA9[c][(wave << 4) << 5], 32, lane, 0);
      v16bf b[2];
      b[0] = frag_b(&sB9[c][0], 32, lane, 0);
      b[1] = frag_b(&sB9[c][16 << 5], 32, lane, 0);
#pragma unroll
      for (int t = 0; t < 2; t++)
        acc[t] = __builtin_amdgcn_wmma_f32_16x16x32_bf16(
            false, a, false, b[t], (short)0, acc[t], false, false);
    }
    __syncthreads();
  }

  const int rowBase = m0 + (wave << 4) + ((lane >> 4) << 3);
  const int colBase = lane & 15;
#pragma unroll
  for (int t = 0; t < 2; t++) {
    int col = colBase + (t << 4);
#pragma unroll
    for (int j = 0; j < 8; j++) {
      int row = rowBase + j;
      if (row < NNODES) out1[(long)row * K1D + col] = acc[t][j];
    }
  }
}

// ---------------------------------------------------------------------------
// KAN layer 2 (32 -> 10), scalar: one node per thread.
// ---------------------------------------------------------------------------
__global__ void kan2_kernel(const float* __restrict__ y,
                            const float* __restrict__ bw2, const float* __restrict__ sw2,
                            const float* __restrict__ sc2, float* __restrict__ out) {
  int n = blockIdx.x * blockDim.x + threadIdx.x;
  if (n >= NNODES) return;
  float acc[OUTD];
#pragma unroll
  for (int o = 0; o < OUTD; o++) acc[o] = 0.f;

  for (int i = 0; i < K1D; i++) {
    float x = y[(long)n * K1D + i];
    float sil = x / (1.f + __expf(-x));
    float u = (x + 2.2f) * 2.5f;
    float bas[8];
#pragma unroll
    for (int c = 0; c < 8; c++) bas[c] = bspl3(u - (float)c);
#pragma unroll
    for (int o = 0; o < OUTD; o++) {
      float s = sil * bw2[o * K1D + i];
      float sc = sc2[o * K1D + i];
#pragma unroll
      for (int c = 0; c < 8; c++) s += bas[c] * sw2[(o * K1D + i) * 8 + c] * sc;
      acc[o] += s;
    }
  }
#pragma unroll
  for (int o = 0; o < OUTD; o++) out[(long)n * OUTD + o] = acc[o];
}

// ---------------------------------------------------------------------------
// Launcher
// ---------------------------------------------------------------------------
extern "C" void kernel_launch(void* const* d_in, const int* in_sizes, int n_in,
                              void* d_out, int out_size, void* d_ws, size_t ws_size,
                              hipStream_t stream) {
  (void)in_sizes; (void)n_in; (void)out_size; (void)ws_size;

  const float* x   = (const float*)d_in[0];
  const float* W1  = (const float*)d_in[3];
  const float* b1  = (const float*)d_in[4];
  const float* g1  = (const float*)d_in[5];
  const float* be1 = (const float*)d_in[6];
  const float* W2  = (const float*)d_in[7];
  const float* b2  = (const float*)d_in[8];
  const float* g2  = (const float*)d_in[9];
  const float* be2 = (const float*)d_in[10];
  const float* Wf  = (const float*)d_in[11];
  const float* bfv = (const float*)d_in[12];
  const float* bw1 = (const float*)d_in[13];
  const float* sw1 = (const float*)d_in[14];
  const float* sc1 = (const float*)d_in[15];
  const float* bw2 = (const float*)d_in[16];
  const float* sw2 = (const float*)d_in[17];
  const float* sc2 = (const float*)d_in[18];
  float* out = (float*)d_out;

  char* ws = (char*)d_ws;
  size_t off = 0;
  auto carve = [&](size_t bytes) -> char* {
    char* p = ws + off;
    off = (off + bytes + 255) & ~(size_t)255;
    return p;
  };

  __bf16* xb   = (__bf16*)carve((size_t)NBR * NNODES * FEATD * 2);
  __bf16* w1t  = (__bf16*)carve((size_t)NBR * HIDD * FEATD * 2);
  __bf16* w2t  = (__bf16*)carve((size_t)NBR * HIDD * HIDD * 2);
  __bf16* wft  = (__bf16*)carve((size_t)F1D * CATD * 2);
  __bf16* wk1  = (__bf16*)carve((size_t)K1D * 9 * F1D * 2);
  float*  hbuf = (float*)carve((size_t)NBR * NNODES * HIDD * 4);
  __bf16* h1n  = (__bf16*)carve((size_t)NBR * NNODES * HIDD * 2);
  __bf16* h2n  = (__bf16*)carve((size_t)NBR * NNODES * HIDD * 2);
  float*  fbuf = (float*)carve((size_t)NNODES * F1D * 4);
  float*  o1   = (float*)carve((size_t)NNODES * K1D * 4);
  float*  psum = (float*)carve((size_t)25 * NBR * HIDD * 4);
  float*  psq  = (float*)carve((size_t)25 * NBR * HIDD * 4);
  float*  sums = (float*)carve((size_t)NBR * HIDD * 4);
  float*  sqs  = (float*)carve((size_t)NBR * HIDD * 4);

  const int Ctot = NBR * HIDD;            // 4096
  const float invN = 1.0f / (float)NNODES;

  // ---- precision conversion / weight packing
  {
    long n = (long)NBR * NNODES * FEATD;
    cvt_bf16_kernel<<<(unsigned)((n + 255) / 256), 256, 0, stream>>>(x, xb, n);
  }
  transpose_cvt_kernel<<<(NBR * FEATD * HIDD + 255) / 256, 256, 0, stream>>>(W1, w1t, NBR, FEATD, HIDD);
  transpose_cvt_kernel<<<(NBR * HIDD * HIDD + 255) / 256, 256, 0, stream>>>(W2, w2t, NBR, HIDD, HIDD);
  transpose_cvt_kernel<<<(CATD * F1D + 255) / 256, 256, 0, stream>>>(Wf, wft, 1, CATD, F1D);
  build_wk1_kernel<<<(K1D * 9 * F1D + 255) / 256, 256, 0, stream>>>(bw1, sw1, sc1, wk1);

  // ---- stage 1: h = x @ W1 (+b1), batchnorm, relu -> h1n (bf16)
  gemm_bf16_kernel<<<dim3(313, HIDD / 128, NBR), 256, 0, stream>>>(
      xb, (long)NNODES * FEATD, FEATD, FEATD, 0L,
      w1t, (long)HIDD * FEATD, FEATD,
      b1, HIDD,
      hbuf, (long)NNODES * HIDD, HIDD,
      NNODES, FEATD, 0);
  colstats_kernel<<<dim3(Ctot / 256, 25, 1), 256, 0, stream>>>(hbuf, psum, psq, NNODES, 800, Ctot, HIDD);
  colstats_finalize_kernel<<<Ctot / 256, 256, 0, stream>>>(psum, psq, sums, sqs, Ctot, 25);
  bn_relu_bf16_kernel<<<(unsigned)(((long)NBR * NNODES * HIDD) / 256), 256, 0, stream>>>(
      hbuf, sums, sqs, g1, be1, h1n, NNODES, HIDD, invN);

  // ---- stage 2: h = h1n @ W2 (+b2), batchnorm, relu -> h2n (bf16)
  gemm_bf16_kernel<<<dim3(313, HIDD / 128, NBR), 256, 0, stream>>>(
      h1n, (long)NNODES * HIDD, HIDD, HIDD, 0L,
      w2t, (long)HIDD * HIDD, HIDD,
      b2, HIDD,
      hbuf, (long)NNODES * HIDD, HIDD,
      NNODES, HIDD, 0);
  colstats_kernel<<<dim3(Ctot / 256, 25, 1), 256, 0, stream>>>(hbuf, psum, psq, NNODES, 800, Ctot, HIDD);
  colstats_finalize_kernel<<<Ctot / 256, 256, 0, stream>>>(psum, psq, sums, sqs, Ctot, 25);
  bn_relu_bf16_kernel<<<(unsigned)(((long)NBR * NNODES * HIDD) / 256), 256, 0, stream>>>(
      hbuf, sums, sqs, g2, be2, h2n, NNODES, HIDD, invN);

  // ---- stage 3: f = relu(concat(h2n) @ Wf + bf)  (concat via A addressing)
  gemm_bf16_kernel<<<dim3(313, F1D / 128, 1), 256, 0, stream>>>(
      h2n, 0L, HIDD, HIDD, (long)NNODES * HIDD,
      wft, 0L, CATD,
      bfv, 0L,
      fbuf, 0L, F1D,
      NNODES, CATD, 1);

  // ---- stage 4: KAN layer 1 (fused SiLU + spline expansion WMMA GEMM)
  kan1_kernel<<<313, 128, 0, stream>>>(fbuf, wk1, o1);

  // ---- stage 5: KAN layer 2 (32 -> 10)
  kan2_kernel<<<(NNODES + 255) / 256, 256, 0, stream>>>(o1, bw2, sw2, sc2, out);
}
